// CombinedChannelGRU_59700045414885
// MI455X (gfx1250) — compile-verified, matrix operand order checked by reading, not run on hardware
//
#include <hip/hip_runtime.h>
#include <hip/hip_bf16.h>

#define Tlen 1024
#define Bb   128
#define Hh   256
#define G3   768   // 3*H
#define INP  128
#define OUTn 128
#define NWG_REC 16

typedef __bf16 bf16_t;
typedef bf16_t bf16x16 __attribute__((ext_vector_type(16)));
typedef float  floatx8 __attribute__((ext_vector_type(8)));

union FragU { bf16x16 bf; uint4 u4[2]; };

__device__ __forceinline__ floatx8 wmma_bf16(bf16x16 a, bf16x16 b, floatx8 c) {
  // D = A(16x32 bf16) * B(32x16 bf16) + C(16x16 f32)
  return __builtin_amdgcn_wmma_f32_16x16x32_bf16(false, a, false, b, (short)0, c,
                                                 false, false);
}

__device__ __forceinline__ bf16x16 ldfrag(const bf16_t* p0, const bf16_t* p1) {
  FragU f;
  f.u4[0] = *reinterpret_cast<const uint4*>(p0);
  f.u4[1] = *reinterpret_cast<const uint4*>(p1);
  return f.bf;
}

__device__ __forceinline__ float fast_sigmoid(float x) {
  return 1.0f / (1.0f + __expf(-x));
}
__device__ __forceinline__ float fast_tanh(float x) {
  x = fminf(fmaxf(x, -15.0f), 15.0f);
  float e = __expf(2.0f * x);
  return (e - 1.0f) / (e + 1.0f);
}

// -------------------------------------------------------------------------
__global__ void zero_counters_kernel(unsigned int* c) {
  if (threadIdx.x < 128) c[threadIdx.x] = 0u;
}

__global__ void cvt_f32_bf16_kernel(const float* __restrict__ s,
                                    bf16_t* __restrict__ d, int n) {
  int i = blockIdx.x * 256 + threadIdx.x;
  if (i < n) d[i] = (bf16_t)s[i];
}

// x[B,C,T,F] -> xs[T,B,C*F] bf16
__global__ void pack_x_kernel(const float* __restrict__ x,
                              bf16_t* __restrict__ xs) {
  int i = blockIdx.x * 256 + threadIdx.x;            // ((t*B+b)*C+c)*F+f
  int f = i & 31;
  int c = (i >> 5) & 3;
  int b = (i >> 7) & 127;
  int t = i >> 14;
  xs[i] = (bf16_t)x[(((b * 4 + c) * 1024 + t) * 32) + f];
}

// -------------------------------------------------------------------------
// Y[M,768] = X[M,K](bf16) * W[768,K]^T(bf16) + bias  (fp32 out, NT stores)
// block = 8 waves; wave -> one 16-row tile x four 16-col tiles.
// The block's 64xK W slice (contiguous in memory) is staged to LDS once via
// async global->LDS (ASYNCcnt), then B fragments come from LDS.
__global__ void __launch_bounds__(256)
wmma_gemm_bias_kernel(const bf16_t* __restrict__ X, const bf16_t* __restrict__ W,
                      const float* __restrict__ bias, float* __restrict__ Y,
                      int K, int mBlocks) {
  __shared__ __align__(16) bf16_t sB[64 * 256];      // up to 32 KB
  const int tid    = threadIdx.x;
  const int wid    = tid >> 5;
  const int lane   = tid & 31;
  const int nIdx   = lane & 15;
  const int laneHi = lane >> 4;
  const int kb     = laneHi * 8;
  const int mb = blockIdx.x % mBlocks;
  const int ng = blockIdx.x / mBlocks;
  const int rowBase = (mb * 8 + wid) * 16;
  const int nBase   = ng * 64;

  // ---- async stage of W rows [nBase, nBase+64) x K into LDS --------------
  {
    const int chunks = (64 * K) >> 3;                // 16-byte chunks
    const unsigned sBase = (unsigned)(unsigned long long)(void*)sB;
    const unsigned long long wBase = (unsigned long long)W;
    const unsigned gBase = (unsigned)((size_t)nBase * K * sizeof(bf16_t));
    for (int idx = tid; idx < chunks; idx += 256) {
      unsigned lOff = sBase + (unsigned)idx * 16u;
      unsigned gOff = gBase + (unsigned)idx * 16u;
      asm volatile("global_load_async_to_lds_b128 %0, %1, %2"
                   :: "v"(lOff), "v"(gOff), "s"(wBase)
                   : "memory");
    }
    asm volatile("s_wait_asynccnt 0x0" ::: "memory");
  }
  __syncthreads();

  floatx8 acc[4];
#pragma unroll
  for (int j = 0; j < 4; ++j) {
    float bv = bias[nBase + j * 16 + nIdx];
#pragma unroll
    for (int r = 0; r < 8; ++r) acc[j][r] = bv;
  }

  const bf16_t* aBase = X + (size_t)(rowBase + nIdx) * K + kb;
  const int kSteps = K >> 5;
  for (int kt = 0; kt < kSteps; ++kt) {
    const bf16_t* ap = aBase + kt * 32;
    bf16x16 a = ldfrag(ap, ap + 16);
#pragma unroll
    for (int j = 0; j < 4; ++j) {
      const bf16_t* bp = sB + (size_t)(j * 16 + nIdx) * K + kt * 32 + kb;
      bf16x16 bfr = ldfrag(bp, bp + 16);
      acc[j] = wmma_bf16(a, bfr, acc[j]);
    }
  }
#pragma unroll
  for (int j = 0; j < 4; ++j)
#pragma unroll
    for (int r = 0; r < 8; ++r) {
      int row = rowBase + r + 8 * laneHi;
      __builtin_nontemporal_store(acc[j][r],
          Y + (size_t)row * G3 + nBase + j * 16 + nIdx);
    }
}

// -------------------------------------------------------------------------
__device__ __forceinline__ void grid_barrier(unsigned int* cnt, unsigned int target) {
  __threadfence();
  __syncthreads();
  if (threadIdx.x == 0) {
    __hip_atomic_fetch_add(cnt, 1u, __ATOMIC_RELEASE, __HIP_MEMORY_SCOPE_AGENT);
    unsigned int v = __hip_atomic_load(cnt, __ATOMIC_ACQUIRE, __HIP_MEMORY_SCOPE_AGENT);
    while (v < target) {
      __builtin_amdgcn_s_sleep(2);
      v = __hip_atomic_load(cnt, __ATOMIC_ACQUIRE, __HIP_MEMORY_SCOPE_AGENT);
    }
  }
  __syncthreads();
}

// Persistent GRU recurrence: 16 WGs x 8 waves. WG owns 16 hidden cols,
// wave owns 16 batch rows. Whh slice (48 rows x 256) staged in LDS once.
// Hidden state double-buffered bf16 in global; per-wave h block in fp32 regs.
// xg for step t+1 is preloaded into registers BEFORE the step-t barrier so
// its HBM latency hides under the barrier spin; only the h-fragment loads
// (true dependency, L2-resident) remain on the serial path.
__global__ void __launch_bounds__(256, 1)
gru_recurrence_kernel(const float* __restrict__ xg,   // [T,B,768], includes bih
                      const float* __restrict__ bhh,  // [768]
                      const bf16_t* __restrict__ Whh, // [768,256] bf16
                      const float* __restrict__ h0,   // [B,256] fp32
                      bf16_t* __restrict__ hbuf0, bf16_t* __restrict__ hbuf1,
                      bf16_t* __restrict__ y,         // [T,B,256] or nullptr
                      unsigned int* __restrict__ counter) {
  __shared__ __align__(16) bf16_t sW[48 * 256];
  const int tid    = threadIdx.x;
  const int wid    = tid >> 5;
  const int lane   = tid & 31;
  const int nIdx   = lane & 15;
  const int laneHi = lane >> 4;
  const int kb     = laneHi * 8;
  const int c0     = blockIdx.x * 16;
  const int rowBase = wid * 16;

  // stage Whh rows for (r,z,n) gate columns c0..c0+15 into LDS
  {
    const uint4* src = reinterpret_cast<const uint4*>(Whh);
    uint4* dst = reinterpret_cast<uint4*>(sW);
    for (int idx = tid; idx < 48 * 32; idx += 256) {
      int lr = idx >> 5, w = idx & 31;
      int g = (lr >> 4) * Hh + c0 + (lr & 15);
      dst[idx] = src[g * 32 + w];
    }
  }

  // init fp32 register block + publish bf16 h to hbuf0
  float hreg[8];
#pragma unroll
  for (int r = 0; r < 8; ++r) {
    int row = rowBase + r + 8 * laneHi;
    float v = h0[row * Hh + c0 + nIdx];
    hreg[r] = v;
    hbuf0[row * Hh + c0 + nIdx] = (bf16_t)v;
  }

  // preload xg for t=0
  const size_t xgLaneOff = (size_t)(rowBase + 8 * laneHi) * G3 + c0 + nIdx;
  float xr[8], xz[8], xn[8];
  {
    const float* p = xg + xgLaneOff;
#pragma unroll
    for (int r = 0; r < 8; ++r) {
      xr[r] = p[r * G3];
      xz[r] = p[r * G3 + Hh];
      xn[r] = p[r * G3 + 2 * Hh];
    }
  }

  __syncthreads();
  grid_barrier(counter, NWG_REC);   // initial h visible everywhere

  const float bR = bhh[c0 + nIdx];
  const float bZ = bhh[Hh + c0 + nIdx];
  const float bN = bhh[2 * Hh + c0 + nIdx];
  const bf16_t* sR = sW + (size_t)nIdx * 256;
  const bf16_t* sZ = sW + (size_t)(16 + nIdx) * 256;
  const bf16_t* sN = sW + (size_t)(32 + nIdx) * 256;
  const int aRow = rowBase + nIdx;

  for (int t = 0; t < Tlen; ++t) {
    const bf16_t* hcur = (t & 1) ? hbuf1 : hbuf0;
    bf16_t* hnext      = (t & 1) ? hbuf0 : hbuf1;

    floatx8 accR, accZ, accN;
#pragma unroll
    for (int r = 0; r < 8; ++r) {
      accR[r] = bR + xr[r];   // xr includes bih_r from the input GEMM
      accZ[r] = bZ + xz[r];
      accN[r] = bN;           // n-gate: h-side only (xn added after r-gating)
    }

    const bf16_t* aBase = hcur + (size_t)aRow * Hh + kb;
#pragma unroll
    for (int ks = 0; ks < 8; ++ks) {
      bf16x16 a  = ldfrag(aBase + ks * 32, aBase + ks * 32 + 16);
      bf16x16 wr = ldfrag(sR + ks * 32 + kb, sR + ks * 32 + kb + 16);
      bf16x16 wz = ldfrag(sZ + ks * 32 + kb, sZ + ks * 32 + kb + 16);
      bf16x16 wn = ldfrag(sN + ks * 32 + kb, sN + ks * 32 + kb + 16);
      accR = wmma_bf16(a, wr, accR);
      accZ = wmma_bf16(a, wz, accZ);
      accN = wmma_bf16(a, wn, accN);
    }

#pragma unroll
    for (int r = 0; r < 8; ++r) {
      int row = rowBase + r + 8 * laneHi;
      float rg = fast_sigmoid(accR[r]);
      float zg = fast_sigmoid(accZ[r]);
      float ng = fast_tanh(xn[r] + rg * accN[r]);
      float hv = (1.0f - zg) * ng + zg * hreg[r];
      hreg[r] = hv;
      bf16_t hb = (bf16_t)hv;
      hnext[row * Hh + c0 + nIdx] = hb;
      if (y) y[((size_t)t * Bb + row) * Hh + c0 + nIdx] = hb;
    }

    // preload next step's xg before the barrier: latency hides in the spin
    if (t + 1 < Tlen) {
      const float* p = xg + (size_t)(t + 1) * Bb * G3 + xgLaneOff;
#pragma unroll
      for (int r = 0; r < 8; ++r) {
        xr[r] = p[r * G3];
        xz[r] = p[r * G3 + Hh];
        xn[r] = p[r * G3 + 2 * Hh];
      }
    }

    grid_barrier(counter, (unsigned int)(NWG_REC * (t + 2)));
  }
}

// -------------------------------------------------------------------------
// out[128,128] = h1[128,256]*fc_w[128,256]^T + fc_b ; one block, 8 waves
__global__ void __launch_bounds__(256)
fc_kernel(const bf16_t* __restrict__ h1, const bf16_t* __restrict__ fcw,
          const float* __restrict__ fcb, float* __restrict__ out) {
  const int wid    = threadIdx.x >> 5;
  const int lane   = threadIdx.x & 31;
  const int nIdx   = lane & 15;
  const int laneHi = lane >> 4;
  const int kb     = laneHi * 8;
  const int rowBase = wid * 16;

  floatx8 acc[8];
#pragma unroll
  for (int j = 0; j < 8; ++j) {
    float bv = fcb[j * 16 + nIdx];
#pragma unroll
    for (int r = 0; r < 8; ++r) acc[j][r] = bv;
  }
  const bf16_t* aBase = h1 + (size_t)(rowBase + nIdx) * Hh + kb;
#pragma unroll
  for (int kt = 0; kt < 8; ++kt) {
    bf16x16 a = ldfrag(aBase + kt * 32, aBase + kt * 32 + 16);
#pragma unroll
    for (int j = 0; j < 8; ++j) {
      const bf16_t* bp = fcw + (size_t)(j * 16 + nIdx) * Hh + kt * 32 + kb;
      bf16x16 b = ldfrag(bp, bp + 16);
      acc[j] = wmma_bf16(a, b, acc[j]);
    }
  }
#pragma unroll
  for (int j = 0; j < 8; ++j)
#pragma unroll
    for (int r = 0; r < 8; ++r)
      out[(size_t)(rowBase + r + 8 * laneHi) * OUTn + j * 16 + nIdx] = acc[j][r];
}

// -------------------------------------------------------------------------
extern "C" void kernel_launch(void* const* d_in, const int* in_sizes, int n_in,
                              void* d_out, int out_size, void* d_ws, size_t ws_size,
                              hipStream_t stream) {
  const float* x    = (const float*)d_in[0];
  const float* h0   = (const float*)d_in[1];
  const float* Wih0 = (const float*)d_in[2];
  const float* Whh0 = (const float*)d_in[3];
  const float* bih0 = (const float*)d_in[4];
  const float* bhh0 = (const float*)d_in[5];
  const float* Wih1 = (const float*)d_in[6];
  const float* Whh1 = (const float*)d_in[7];
  const float* bih1 = (const float*)d_in[8];
  const float* bhh1 = (const float*)d_in[9];
  const float* fcw  = (const float*)d_in[10];
  const float* fcb  = (const float*)d_in[11];
  float* out = (float*)d_out;

  char* ws = (char*)d_ws;
  size_t off = 0;
  auto take = [&](size_t bytes) -> char* {
    char* p = ws + off;
    off += (bytes + 255) & ~(size_t)255;
    return p;
  };
  const size_t TB = (size_t)Tlen * Bb;
  bf16_t* xs    = (bf16_t*)take(TB * INP * 2);         // packed input, bf16
  float*  xgbuf = (float*) take(TB * G3 * 4);          // pre-activations (reused L0/L1)
  bf16_t* y0    = (bf16_t*)take(TB * Hh * 2);          // layer-0 outputs, bf16
  bf16_t* wih0b = (bf16_t*)take((size_t)G3 * INP * 2);
  bf16_t* whh0b = (bf16_t*)take((size_t)G3 * Hh * 2);
  bf16_t* wih1b = (bf16_t*)take((size_t)G3 * Hh * 2);
  bf16_t* whh1b = (bf16_t*)take((size_t)G3 * Hh * 2);
  bf16_t* wfcbf = (bf16_t*)take((size_t)OUTn * Hh * 2);
  bf16_t* h0a   = (bf16_t*)take((size_t)Bb * Hh * 2);
  bf16_t* h0b   = (bf16_t*)take((size_t)Bb * Hh * 2);
  bf16_t* h1a   = (bf16_t*)take((size_t)Bb * Hh * 2);
  bf16_t* h1b   = (bf16_t*)take((size_t)Bb * Hh * 2);
  unsigned int* cnts = (unsigned int*)take(512);       // two counters, separate lines

  zero_counters_kernel<<<1, 128, 0, stream>>>(cnts);
  pack_x_kernel<<<dim3((unsigned)((TB * INP) / 256)), 256, 0, stream>>>(x, xs);
  auto cvt = [&](const float* s, bf16_t* d, int n) {
    cvt_f32_bf16_kernel<<<dim3((unsigned)((n + 255) / 256)), 256, 0, stream>>>(s, d, n);
  };
  cvt(Wih0, wih0b, G3 * INP);
  cvt(Whh0, whh0b, G3 * Hh);
  cvt(Wih1, wih1b, G3 * Hh);
  cvt(Whh1, whh1b, G3 * Hh);
  cvt(fcw,  wfcbf, OUTn * Hh);

  const int mBlocks = (int)(TB / 128);                 // 1024
  // layer 0: xg = xs * Wih0^T + bih0
  wmma_gemm_bias_kernel<<<dim3((unsigned)(mBlocks * 12)), 256, 0, stream>>>(
      xs, wih0b, bih0, xgbuf, INP, mBlocks);
  gru_recurrence_kernel<<<NWG_REC, 256, 0, stream>>>(
      xgbuf, bhh0, whh0b, h0, h0a, h0b, y0, cnts + 0);
  // layer 1: xg = y0 * Wih1^T + bih1
  wmma_gemm_bias_kernel<<<dim3((unsigned)(mBlocks * 12)), 256, 0, stream>>>(
      y0, wih1b, bih1, xgbuf, Hh, mBlocks);
  gru_recurrence_kernel<<<NWG_REC, 256, 0, stream>>>(
      xgbuf, bhh1, whh1b, h0 + Bb * Hh, h1a, h1b, nullptr, cnts + 64);
  // T=1024 even -> final hidden state lives in h1a
  fc_kernel<<<1, 256, 0, stream>>>(h1a, wfcbf, fcb, out);
}
